// YOLOTOS_75136157876612
// MI455X (gfx1250) — compile-verified, matrix operand order checked by reading, not run on hardware
//
#include <hip/hip_runtime.h>

// ---------------------------------------------------------------------------
// YOLO-TOS forward for MI455X (gfx1250, wave32, WMMA).
// Persistent cooperative kernel: 16 blocks x 256 threads, grid barrier per
// phase, bf16 WMMA (v_wmma_f32_16x16x32_bf16) for all GEMMs, LDS-resident
// attention K/V (ds_load_b128), exp2-based softmax on v_exp_f32.
// All matrix-fragment loads forced to global AS => global_load_b128 (no flat).
// ---------------------------------------------------------------------------

#define B_    32
#define T_    96
#define H_    512
#define FEAT_ 2048
#define NCP_  82
#define NBP_  226
#define K5_   2560          // 5*H
#define G3_   1536          // 3*H
#define NOUT_ 992           // 82 + 904 + 3 padded to 62*16
#define NWG_  16
#define NTHR_ 256
#define BT_   (B_*T_)

#define GAS __attribute__((address_space(1)))   // global

typedef unsigned int  u32x4   __attribute__((ext_vector_type(4)));
typedef float         f32x4   __attribute__((ext_vector_type(4)));
typedef __bf16        bf16x16 __attribute__((ext_vector_type(16)));
typedef float         f32x8   __attribute__((ext_vector_type(8)));

union FragU { bf16x16 v; u32x4 q[2]; };

__device__ __forceinline__ unsigned short f2bf(float f) {
  unsigned u = __float_as_uint(f);
  return (unsigned short)((u + 0x7FFFu + ((u >> 16) & 1u)) >> 16);  // RNE
}

// One WMMA operand fragment per lane = two contiguous 16B global loads from a
// row-major bf16 row (A: row = matrix row m, B: row = weight row n).
// Per ISA layout: element e of lane l holds k = 32*kc + 16*(e>>3) + 8*(l>>4) + (e&7).
__device__ __forceinline__ bf16x16 ldfrag(const unsigned short* row, int kc, int half) {
  const GAS unsigned short* g = (const GAS unsigned short*)row;
  FragU f;
  f.q[0] = *(const GAS u32x4*)(g + kc * 32 + half * 8);
  f.q[1] = *(const GAS u32x4*)(g + kc * 32 + 16 + half * 8);
  return f.v;
}

__device__ __forceinline__ f32x8 wmma_bf16(bf16x16 a, bf16x16 b, f32x8 c) {
  return __builtin_amdgcn_wmma_f32_16x16x32_bf16(false, a, false, b, (short)0, c, false, false);
}

__device__ __forceinline__ float fexp2(float x) {
#if __has_builtin(__builtin_amdgcn_exp2f)
  return __builtin_amdgcn_exp2f(x);
#else
  return exp2f(x);
#endif
}
__device__ __forceinline__ float frcp(float x) {
#if __has_builtin(__builtin_amdgcn_rcpf)
  return __builtin_amdgcn_rcpf(x);
#else
  return 1.0f / x;
#endif
}
#define L2E_ 1.4426950408889634f
__device__ __forceinline__ float fsig(float x)  { return frcp(1.f + fexp2(-x * L2E_)); }
__device__ __forceinline__ float ftanhf(float x){ return 1.f - 2.f * frcp(fexp2(2.f * L2E_ * x) + 1.f); }

// Monotonic grid barrier (counter zeroed by hipMemsetAsync each launch).
__device__ __forceinline__ void grid_barrier(unsigned* cnt, unsigned goal) {
  __syncthreads();
  if (threadIdx.x == 0) {
    __threadfence();
    atomicAdd(cnt, 1u);
    while (__hip_atomic_load(cnt, __ATOMIC_RELAXED, __HIP_MEMORY_SCOPE_AGENT) < goal)
      __builtin_amdgcn_s_sleep(2);
    __threadfence();
  }
  __syncthreads();
}

// Map packed output column n -> reference tuple layout (oc,ox,oy,ow,oh,od).
__device__ __forceinline__ void store_out(float* out, int t, int b, int n, float v) {
  GAS float* o = (GAS float*)out;
  if (n < 82) {
    o[((size_t)b * T_ + t) * 82 + n] = v;
  } else if (n < 308) {                       // pos[:,0] -> ox
    o[(size_t)BT_ * 82  + ((size_t)b * T_ + t) * 226 + (n - 82)]  = v;
  } else if (n < 534) {                       // pos[:,1] -> oy
    o[(size_t)BT_ * 308 + ((size_t)b * T_ + t) * 226 + (n - 308)] = v;
  } else if (n < 760) {
    // pos[:,2] : discarded (original-code bug preserved)
  } else if (n < 986) {                       // pos[:,3] -> ow
    o[(size_t)BT_ * 534 + ((size_t)b * T_ + t) * 226 + (n - 760)] = v;
  } else if (n < 989) {                       // isdone
    o[(size_t)BT_ * 986 + ((size_t)b * T_ + t) * 3 + (n - 986)] = v;
  }                                           // 989..991 padding
}

// ---------------------------------------------------------------------------
// Pre-kernels
// ---------------------------------------------------------------------------
__global__ void cvt_bf16_k(const float* __restrict__ src, unsigned short* __restrict__ dst, int n) {
  int i = blockIdx.x * blockDim.x + threadIdx.x;
  int st = gridDim.x * blockDim.x;
  for (; i < n; i += st) dst[i] = f2bf(src[i]);
}

__global__ void pack_wcat_k(const float* __restrict__ Wc, const float* __restrict__ Wp,
                            const float* __restrict__ Wd, const float* __restrict__ bc,
                            const float* __restrict__ bp, const float* __restrict__ bd,
                            unsigned short* __restrict__ Wcat, float* __restrict__ bcat) {
  int i = blockIdx.x * blockDim.x + threadIdx.x;
  int st = gridDim.x * blockDim.x;
  for (int idx = i; idx < NOUT_ * H_; idx += st) {
    int row = idx >> 9, col = idx & 511;
    float v;
    if      (row < 82)  v = Wc[(size_t)row * H_ + col];
    else if (row < 986) v = Wp[(size_t)(row - 82) * H_ + col];
    else if (row < 989) v = Wd[(size_t)(row - 986) * H_ + col];
    else                v = 0.f;
    Wcat[idx] = f2bf(v);
  }
  if (i < NOUT_) {
    float v = (i < 82) ? bc[i] : (i < 986) ? bp[i - 82] : (i < 989) ? bd[i - 986] : 0.f;
    bcat[i] = v;
  }
}

// img = image_feat @ Wb^T + bb ; k/v projections ; per-batch kmax/kmin.
__global__ void img_kv_k(const float* __restrict__ imf, const float* __restrict__ Wb,
                         const float* __restrict__ bb, const float* __restrict__ aw,
                         const float* __restrict__ ab, float* __restrict__ kproj,
                         float* __restrict__ vproj, float* __restrict__ kmm) {
  __shared__ float rmax[NTHR_], rmin[NTHR_];
  int b = blockIdx.x, tid = threadIdx.x;
  float lmax = -3.4e38f, lmin = 3.4e38f;
  const f32x4* x4 = (const f32x4*)(imf + (size_t)b * FEAT_);
  for (int jj = tid; jj < H_; jj += NTHR_) {
    const f32x4* w4 = (const f32x4*)(Wb + (size_t)jj * FEAT_);
    float acc = bb[jj];
    for (int kk = 0; kk < FEAT_ / 4; ++kk) {
      f32x4 a = x4[kk], w = w4[kk];
      acc += a[0] * w[0] + a[1] * w[1] + a[2] * w[2] + a[3] * w[3];
    }
    float k_ = acc * aw[1] + ab[1];
    float v_ = acc * aw[2] + ab[2];
    kproj[(size_t)b * H_ + jj] = k_;
    vproj[(size_t)b * H_ + jj] = v_;
    lmax = fmaxf(lmax, k_);
    lmin = fminf(lmin, k_);
  }
  rmax[tid] = lmax; rmin[tid] = lmin;
  __syncthreads();
  for (int s = NTHR_ / 2; s > 0; s >>= 1) {
    if (tid < s) { rmax[tid] = fmaxf(rmax[tid], rmax[tid + s]); rmin[tid] = fminf(rmin[tid], rmin[tid + s]); }
    __syncthreads();
  }
  if (tid == 0) { kmm[b] = rmax[0]; kmm[32 + b] = rmin[0]; }
}

// ---------------------------------------------------------------------------
// Persistent main kernel
// ---------------------------------------------------------------------------
__global__ void __launch_bounds__(NTHR_) yolotos_main(
    const int* __restrict__ yc, const int* __restrict__ yx, const int* __restrict__ yy,
    const int* __restrict__ yw, const int* __restrict__ yh,
    const unsigned short* __restrict__ clsb, const unsigned short* __restrict__ binb,
    const unsigned short* __restrict__ Wihb, const unsigned short* __restrict__ Whhb,
    const unsigned short* __restrict__ Wcatb, const float* __restrict__ bcat,
    const float* __restrict__ b_ih, const float* __restrict__ b_hh,
    const float* __restrict__ kproj, const float* __restrict__ vproj,
    const float* __restrict__ kmm,
    const float* __restrict__ aw, const float* __restrict__ ab,
    const float* __restrict__ owp, const float* __restrict__ obp,
    float* __restrict__ hf, unsigned short* __restrict__ hb,
    float* __restrict__ qbuf, unsigned* __restrict__ barcnt,
    float* __restrict__ out) {
  extern __shared__ float smem[];
  float* sk  = smem;                 // [32][512] k_proj (step-invariant)
  float* sv  = smem + B_ * H_;       // [32][512] v_proj
  float* skm = sv   + B_ * H_;       // [64] kmax(32) | kmin(32)

  const int tid  = threadIdx.x;
  const int lane = tid & 31;
  const int wid  = tid >> 5;
  const int blk  = blockIdx.x;
  const int gtid = blk * NTHR_ + tid;

  for (int i = tid; i < B_ * H_; i += NTHR_) { sk[i] = kproj[i]; sv[i] = vproj[i]; }
  for (int i = tid; i < 64; i += NTHR_) skm[i] = kmm[i];
  __syncthreads();

  const float w_q = aw[0], b_q = ab[0];
  const float w_o = owp[0], b_o = obp[0];

  // ---- phase-G (gates) lane-invariant setup: waves 0..1 of each block ----
  const int nloc = lane & 15;        // tile column / A-row within tile
  const int half = lane >> 4;
  const int jt   = blk * 2 + wid;    // hidden-unit tile (valid when wid<2)
  const int jcol = jt * 16 + nloc;   // hidden unit handled by this lane
  const unsigned short* wr_r = Wihb + (size_t)(         jcol) * K5_;
  const unsigned short* wr_z = Wihb + (size_t)(H_     + jcol) * K5_;
  const unsigned short* wr_n = Wihb + (size_t)(2 * H_ + jcol) * K5_;
  const unsigned short* hr_r = Whhb + (size_t)(         jcol) * H_;
  const unsigned short* hr_z = Whhb + (size_t)(H_     + jcol) * H_;
  const unsigned short* hr_n = Whhb + (size_t)(2 * H_ + jcol) * H_;
  float bias_r = 0.f, bias_z = 0.f, bias_in = 0.f, bias_hn = 0.f;
  if (wid < 2) {
    bias_r  = b_ih[jcol]          + b_hh[jcol];
    bias_z  = b_ih[H_ + jcol]     + b_hh[H_ + jcol];
    bias_in = b_ih[2 * H_ + jcol];
    bias_hn = b_hh[2 * H_ + jcol];
  }
  const int bm0 = nloc, bm1 = nloc + 16;     // batch rows of the two M-tiles
  const unsigned short* hrow0 = hb + (size_t)bm0 * H_;
  const unsigned short* hrow1 = hb + (size_t)bm1 * H_;

  unsigned barid = 0;

#pragma unroll 1
  for (int t = 0; t < T_; ++t) {
    // ================= Phase G : GRU gates (32 waves) =================
    if (wid < 2) {
      const unsigned short* a0seg[5];
      const unsigned short* a1seg[5];
      if (t == 0) {
        a0seg[0] = clsb; a1seg[0] = clsb;
        for (int s = 1; s < 5; ++s) { a0seg[s] = binb; a1seg[s] = binb; }
      } else {
        a0seg[0] = clsb + (size_t)yc[bm0 * T_ + t - 1] * H_;
        a1seg[0] = clsb + (size_t)yc[bm1 * T_ + t - 1] * H_;
        a0seg[1] = binb + (size_t)yx[bm0 * T_ + t - 1] * H_;
        a1seg[1] = binb + (size_t)yx[bm1 * T_ + t - 1] * H_;
        a0seg[2] = binb + (size_t)yy[bm0 * T_ + t - 1] * H_;
        a1seg[2] = binb + (size_t)yy[bm1 * T_ + t - 1] * H_;
        a0seg[3] = binb + (size_t)yw[bm0 * T_ + t - 1] * H_;
        a1seg[3] = binb + (size_t)yw[bm1 * T_ + t - 1] * H_;
        a0seg[4] = binb + (size_t)yh[bm0 * T_ + t - 1] * H_;
        a1seg[4] = binb + (size_t)yh[bm1 * T_ + t - 1] * H_;
      }

      f32x8 ar0 = (f32x8)0.f, ar1 = (f32x8)0.f;   // r gate (gi+gh fused)
      f32x8 az0 = (f32x8)0.f, az1 = (f32x8)0.f;   // z gate (gi+gh fused)
      f32x8 ai0 = (f32x8)0.f, ai1 = (f32x8)0.f;   // inn (input n)
      f32x8 ah0 = (f32x8)0.f, ah1 = (f32x8)0.f;   // hn  (hidden n)

      // gi : K = 2560 over 5 embedding segments
#pragma unroll 1
      for (int s = 0; s < 5; ++s) {
        const unsigned short* rA0 = a0seg[s];
        const unsigned short* rA1 = a1seg[s];
        const unsigned short* rBr = wr_r + s * 512;
        const unsigned short* rBz = wr_z + s * 512;
        const unsigned short* rBn = wr_n + s * 512;
#pragma unroll 2
        for (int kl = 0; kl < 16; ++kl) {
          __builtin_prefetch(rBr + (kl + 2) * 32, 0, 1);
          __builtin_prefetch(rBn + (kl + 2) * 32, 0, 1);
          bf16x16 A0 = ldfrag(rA0, kl, half);
          bf16x16 A1 = ldfrag(rA1, kl, half);
          bf16x16 Br = ldfrag(rBr, kl, half);
          bf16x16 Bz = ldfrag(rBz, kl, half);
          bf16x16 Bn = ldfrag(rBn, kl, half);
          ar0 = wmma_bf16(A0, Br, ar0); ar1 = wmma_bf16(A1, Br, ar1);
          az0 = wmma_bf16(A0, Bz, az0); az1 = wmma_bf16(A1, Bz, az1);
          ai0 = wmma_bf16(A0, Bn, ai0); ai1 = wmma_bf16(A1, Bn, ai1);
        }
      }
      // gh : K = 512 from previous h (bf16, L2-resident)
#pragma unroll 2
      for (int kl = 0; kl < 16; ++kl) {
        bf16x16 A0 = ldfrag(hrow0, kl, half);
        bf16x16 A1 = ldfrag(hrow1, kl, half);
        bf16x16 Br = ldfrag(hr_r, kl, half);
        bf16x16 Bz = ldfrag(hr_z, kl, half);
        bf16x16 Bn = ldfrag(hr_n, kl, half);
        ar0 = wmma_bf16(A0, Br, ar0); ar1 = wmma_bf16(A1, Br, ar1);
        az0 = wmma_bf16(A0, Bz, az0); az1 = wmma_bf16(A1, Bz, az1);
        ah0 = wmma_bf16(A0, Bn, ah0); ah1 = wmma_bf16(A1, Bn, ah1);
      }

      // GRU combine + q projection. D element r of lane: m = r + 8*half.
      const GAS float* hfg = (const GAS float*)hf;
      GAS float* qg = (GAS float*)qbuf;
#pragma unroll
      for (int r = 0; r < 8; ++r) {
        {
          int brow = half * 8 + r;                       // m-tile 0
          float R = fsig(ar0[r] + bias_r);
          float Z = fsig(az0[r] + bias_z);
          float N = ftanhf(ai0[r] + bias_in + R * (ah0[r] + bias_hn));
          float hg = (1.f - Z) * N + Z * hfg[(size_t)brow * H_ + jcol];
          qg[(size_t)brow * H_ + jcol] = hg * w_q + b_q;
        }
        {
          int brow = 16 + half * 8 + r;                  // m-tile 1
          float R = fsig(ar1[r] + bias_r);
          float Z = fsig(az1[r] + bias_z);
          float N = ftanhf(ai1[r] + bias_in + R * (ah1[r] + bias_hn));
          float hg = (1.f - Z) * N + Z * hfg[(size_t)brow * H_ + jcol];
          qg[(size_t)brow * H_ + jcol] = hg * w_q + b_q;
        }
      }
    }
    grid_barrier(barcnt, ++barid * NWG_);

    // ============ Phase A : scalar-embed attention (all 128 waves) ============
#pragma unroll 1
    for (int p = 0; p < 4; ++p) {
      int pid = p * (NWG_ * NTHR_) + gtid;   // pid = b*512 + i
      int b   = pid >> 9;
      float qs = qbuf[pid] * L2E_;
      float mm = (qs > 0.f) ? qs * skm[b] : qs * skm[32 + b];
      const f32x4* kr4 = (const f32x4*)(sk + (size_t)b * H_);   // ds_load_b128
      const f32x4* vr4 = (const f32x4*)(sv + (size_t)b * H_);
      float se = 0.f, wv = 0.f;
#pragma unroll 4
      for (int jj = 0; jj < H_ / 4; ++jj) {
        f32x4 k4 = kr4[jj];
        f32x4 v4 = vr4[jj];
        float e0 = fexp2(qs * k4[0] - mm);   // v_exp_f32
        float e1 = fexp2(qs * k4[1] - mm);
        float e2 = fexp2(qs * k4[2] - mm);
        float e3 = fexp2(qs * k4[3] - mm);
        se += (e0 + e1) + (e2 + e3);
        wv += (e0 * v4[0] + e1 * v4[1]) + (e2 * v4[2] + e3 * v4[3]);
      }
      float hatt = (wv * frcp(se)) * w_o + b_o;
      hf[pid] = hatt;
      hb[pid] = f2bf(hatt);
    }
    grid_barrier(barcnt, ++barid * NWG_);

    // ============ Phase O : output heads (62 waves, WMMA) ============
    {
      int wglobal = blk * 8 + wid;
      if (wglobal < NOUT_ / 16) {
        int nt = wglobal;
        const unsigned short* wrow = Wcatb + (size_t)(nt * 16 + nloc) * H_;
        float bo = bcat[nt * 16 + nloc];
        f32x8 c0 = (f32x8)0.f, c1 = (f32x8)0.f;
#pragma unroll 2
        for (int kl = 0; kl < 16; ++kl) {
          bf16x16 A0 = ldfrag(hb + (size_t)nloc * H_, kl, half);
          bf16x16 A1 = ldfrag(hb + (size_t)(nloc + 16) * H_, kl, half);
          bf16x16 Bw = ldfrag(wrow, kl, half);
          c0 = wmma_bf16(A0, Bw, c0);
          c1 = wmma_bf16(A1, Bw, c1);
        }
        int ncol = nt * 16 + nloc;
#pragma unroll
        for (int r = 0; r < 8; ++r) {
          store_out(out, t, half * 8 + r,      ncol, c0[r] + bo);
          store_out(out, t, 16 + half * 8 + r, ncol, c1[r] + bo);
        }
      }
    }
  }
}

// ---------------------------------------------------------------------------
// Host launcher
// ---------------------------------------------------------------------------
extern "C" void kernel_launch(void* const* d_in, const int* in_sizes, int n_in,
                              void* d_out, int out_size, void* d_ws, size_t ws_size,
                              hipStream_t stream) {
  (void)in_sizes; (void)n_in; (void)out_size; (void)ws_size;
  const float* image_feat = (const float*)d_in[0];
  const int*   yc = (const int*)d_in[1];
  const int*   yx = (const int*)d_in[2];
  const int*   yy = (const int*)d_in[3];
  const int*   yw = (const int*)d_in[4];
  const int*   yh = (const int*)d_in[5];
  /* d_in[6] y_isdone unused */
  const float* class_emb = (const float*)d_in[7];
  const float* bin_emb   = (const float*)d_in[8];
  const float* Wb   = (const float*)d_in[9];
  const float* bb   = (const float*)d_in[10];
  const float* W_ih = (const float*)d_in[11];
  const float* b_ih = (const float*)d_in[12];
  const float* W_hh = (const float*)d_in[13];
  const float* b_hh = (const float*)d_in[14];
  const float* aw   = (const float*)d_in[15];
  const float* ab   = (const float*)d_in[16];
  const float* owp  = (const float*)d_in[17];
  const float* obp  = (const float*)d_in[18];
  const float* Wc   = (const float*)d_in[19];
  const float* bc   = (const float*)d_in[20];
  const float* Wp   = (const float*)d_in[21];
  const float* bp   = (const float*)d_in[22];
  const float* Wd   = (const float*)d_in[23];
  const float* bd   = (const float*)d_in[24];
  float* out = (float*)d_out;

  // ---- carve workspace (256B aligned) ----
  char* w = (char*)d_ws;
  size_t off = 0;
  auto carve = [&](size_t bytes) -> void* {
    void* p = w + off;
    off = (off + bytes + 255) & ~(size_t)255;
    return p;
  };
  unsigned short* Wihb  = (unsigned short*)carve((size_t)G3_ * K5_ * 2);
  unsigned short* Whhb  = (unsigned short*)carve((size_t)G3_ * H_ * 2);
  unsigned short* Wcatb = (unsigned short*)carve((size_t)NOUT_ * H_ * 2);
  unsigned short* clsb  = (unsigned short*)carve((size_t)NCP_ * H_ * 2);
  unsigned short* binb  = (unsigned short*)carve((size_t)NBP_ * H_ * 2);
  float* bcat  = (float*)carve(NOUT_ * 4);
  float* kproj = (float*)carve((size_t)B_ * H_ * 4);
  float* vproj = (float*)carve((size_t)B_ * H_ * 4);
  float* kmm   = (float*)carve(64 * 4);
  float* hf    = (float*)carve((size_t)B_ * H_ * 4);
  unsigned short* hb = (unsigned short*)carve((size_t)B_ * H_ * 2);
  float* qbuf  = (float*)carve((size_t)B_ * H_ * 4);
  unsigned* barcnt = (unsigned*)carve(256);

  // ---- init state (graph-capture-safe async memsets) ----
  hipMemsetAsync(barcnt, 0, 256, stream);
  hipMemsetAsync(hf, 0, (size_t)B_ * H_ * 4, stream);
  hipMemsetAsync(hb, 0, (size_t)B_ * H_ * 2, stream);
  // oh output (B,T,226) is identically zero in the reference
  hipMemsetAsync((char*)out + (size_t)BT_ * 760 * 4, 0, (size_t)BT_ * 226 * 4, stream);

  // ---- precompute: bf16 weights/embeddings, packed heads, img/k/v ----
  cvt_bf16_k<<<512, 256, 0, stream>>>(W_ih, Wihb, G3_ * K5_);
  cvt_bf16_k<<<256, 256, 0, stream>>>(W_hh, Whhb, G3_ * H_);
  cvt_bf16_k<<<32, 256, 0, stream>>>(class_emb, clsb, NCP_ * H_);
  cvt_bf16_k<<<64, 256, 0, stream>>>(bin_emb, binb, NBP_ * H_);
  pack_wcat_k<<<256, 256, 0, stream>>>(Wc, Wp, Wd, bc, bp, bd, Wcatb, bcat);
  img_kv_k<<<B_, NTHR_, 0, stream>>>(image_feat, Wb, bb, aw, ab, kproj, vproj, kmm);

  // ---- persistent recurrent kernel ----
  size_t smem_bytes = (size_t)(2 * B_ * H_ + 64) * sizeof(float);  // 131,328 B < 320KB LDS
  yolotos_main<<<NWG_, NTHR_, smem_bytes, stream>>>(
      yc, yx, yy, yw, yh, clsb, binb, Wihb, Whhb, Wcatb, bcat, b_ih, b_hh,
      kproj, vproj, kmm, aw, ab, owp, obp, hf, hb, qbuf, barcnt, out);
}
